// Decoder_27041114096334
// MI455X (gfx1250) — matrix operand order repaired and checked by reading, not compiled, not run on hardware
//
#include <hip/hip_runtime.h>
#include <hip/hip_bf16.h>
#include <math.h>

// ---------------- problem constants ----------------
#define BB      32
#define T_ENC   256
#define T_OUT   400
#define N_MEL   80
#define EMB     512
#define RNN     1024      // ATT_RNN == DEC_RNN
#define PRENETD 256
#define ATT_DIM 128
#define N_FILT  32
#define KSIZE   31
#define NG      4096              // 4*RNN gate width
#define KATT    (PRENETD + EMB + RNN)   // 1792  [x | ctx | ah]   (7  x 256)
#define KDEC    (RNN + EMB + RNN)       // 2560  [ah | ctx | dh]  (10 x 256)
#define KCHUNK  256                     // K staging chunk (32x256 bf16 = 16KB LDS)

typedef __bf16 bf16_t;
typedef __attribute__((ext_vector_type(16))) __bf16 v16bf;
typedef __attribute__((ext_vector_type(8)))  float  v8f;

__device__ __forceinline__ float sigmoidf_(float x) { return 1.0f / (1.0f + expf(-x)); }

// ============================================================
// Phase-A kernels (run once, fully parallel)
// ============================================================

__global__ void zero_words_kernel(unsigned int* __restrict__ p, size_t n) {
  size_t i = (size_t)blockIdx.x * blockDim.x + threadIdx.x;
  if (i < n) p[i] = 0u;
}

// pack [wih | whh] -> bf16 row-major (NG x K), fuse biases
__global__ void prep_w_kernel(const float* __restrict__ wih, const float* __restrict__ whh,
                              const float* __restrict__ bih, const float* __restrict__ bhh,
                              bf16_t* __restrict__ W, float* __restrict__ bias,
                              int k_ih, int K) {
  size_t idx = (size_t)blockIdx.x * blockDim.x + threadIdx.x;
  size_t total = (size_t)NG * (size_t)K;
  if (idx < total) {
    int n = (int)(idx / (size_t)K);
    int k = (int)(idx % (size_t)K);
    float v = (k < k_ih) ? wih[(size_t)n * k_ih + k]
                         : whh[(size_t)n * RNN + (k - k_ih)];
    W[idx] = (bf16_t)v;
  }
  if (idx < NG) bias[idx] = bih[idx] + bhh[idx];
}

// prenet for all timesteps (hoisted out of the serial loop), bf16 output
__global__ void prenet_kernel(const float* __restrict__ dec_inputs,
                              const float* __restrict__ w1, const float* __restrict__ w2,
                              bf16_t* __restrict__ xs_bf) {
  const int b = blockIdx.x & (BB - 1);
  const int t = blockIdx.x >> 5;           // 0..T_OUT-1
  const int j = threadIdx.x;               // 0..255
  __shared__ float h1[PRENETD];
  float acc = 0.0f;
  if (t > 0) {
    const float* x = dec_inputs + ((size_t)b * T_OUT + (t - 1)) * N_MEL;
    const float* wr = w1 + (size_t)j * N_MEL;
    for (int m = 0; m < N_MEL; ++m) acc += x[m] * wr[m];
  }
  h1[j] = fmaxf(acc, 0.0f);
  __syncthreads();
  const float* wr2 = w2 + (size_t)j * PRENETD;
  float acc2 = 0.0f;
  for (int m = 0; m < PRENETD; ++m) acc2 += h1[m] * wr2[m];
  xs_bf[((size_t)t * BB + b) * PRENETD + j] = (bf16_t)fmaxf(acc2, 0.0f);
}

// keys = memory @ wk^T
__global__ void keys_kernel(const float* __restrict__ memory, const float* __restrict__ wk,
                            float* __restrict__ keys) {
  const int bt = blockIdx.x;               // 0..B*T_ENC-1
  const int d  = threadIdx.x;              // 0..127
  const float* mrow = memory + (size_t)bt * EMB;
  const float* wr   = wk + (size_t)d * EMB;
  float acc = 0.0f;
  for (int k = 0; k < EMB; ++k) acc += mrow[k] * wr[k];
  keys[(size_t)bt * ATT_DIM + d] = acc;
}

// ============================================================
// Per-step kernels (400 sequential iterations)
// ============================================================

__global__ void copy_x_kernel(const bf16_t* __restrict__ xs_bf, bf16_t* __restrict__ A_att,
                              int tstep) {
  const int idx = blockIdx.x * blockDim.x + threadIdx.x;   // 0..BB*PRENETD-1
  const int b = idx >> 8, j = idx & 255;
  A_att[(size_t)b * KATT + j] = xs_bf[((size_t)tstep * BB + b) * PRENETD + j];
}

// ------------------------------------------------------------
// WMMA GEMM: gates[32, 4096] = A[32, K](bf16) @ W[4096, K]^T(bf16) + bias
//  - A panel staged into LDS via async global->LDS DMA, double-buffered:
//    DMA of chunk i+1 overlaps WMMA on chunk i (ASYNCcnt + barrier discipline)
//  - one wave owns a 32(M) x 16(N) tile -> 2x v_wmma_f32_16x16x32_bf16 / K-step
//  - weight stream (B) read directly from global (L2-resident), with
//    global_prefetch one chunk ahead
// ------------------------------------------------------------
__global__ void gemm_gates_kernel(const bf16_t* __restrict__ A, const bf16_t* __restrict__ W,
                                  const float* __restrict__ bias, float* __restrict__ out,
                                  int K) {
  __shared__ __align__(16) bf16_t sA[2][32 * KCHUNK];   // 2 x 16KB

  const int tid  = threadIdx.x;
  const int lane = tid & 31;
  const int wid  = tid >> 5;                     // 0..7
  const int n0   = blockIdx.x * 128 + wid * 16;  // 16-wide N tile
  const int col  = lane & 15;
  const int half = lane >> 4;

  // LDS byte offsets of the two staging buffers (generic ptr low 32 bits == LDS addr)
  const unsigned lds0 = (unsigned)(uintptr_t)(void*)&sA[0][0];
  const unsigned lds1 = (unsigned)(uintptr_t)(void*)&sA[1][0];

  // staging assignment: 8 threads per row, 32 bf16 (64B = 4 x b128) per thread
  const int srow = tid >> 3;             // 0..31
  const int sseg = (tid & 7) * 32;       // element offset within row chunk

  const float bcol = bias[n0 + col];
  v8f c0, c1;
#pragma unroll
  for (int i = 0; i < 8; ++i) { c0[i] = bcol; c1[i] = bcol; }

  const bf16_t* wrow = W + (size_t)(n0 + col) * K;   // weight row = output column

  // issue async stage of chunk 0 into buffer 0
  {
    const bf16_t* gsrc = A + (size_t)srow * K + sseg;
    const unsigned ldst = lds0 + (unsigned)(srow * KCHUNK + sseg) * 2u;
    asm volatile(
        "global_load_async_to_lds_b128 %0, %1, off\n\t"
        "global_load_async_to_lds_b128 %0, %1, off offset:16\n\t"
        "global_load_async_to_lds_b128 %0, %1, off offset:32\n\t"
        "global_load_async_to_lds_b128 %0, %1, off offset:48"
        :: "v"(ldst), "v"(gsrc) : "memory");
  }

  const int nch = K / KCHUNK;
  for (int ch = 0; ch < nch; ++ch) {
    // wait for the stage issued for this chunk, make it visible to all waves
    asm volatile("s_wait_asynccnt 0x0" ::: "memory");
    __syncthreads();

    // kick off DMA of the next chunk into the other buffer (overlaps compute)
    if (ch + 1 < nch) {
      const unsigned ldsn = (ch & 1) ? lds0 : lds1;
      const int kbn = (ch + 1) * KCHUNK;
      const bf16_t* gsrc = A + (size_t)srow * K + kbn + sseg;
      const unsigned ldst = ldsn + (unsigned)(srow * KCHUNK + sseg) * 2u;
      asm volatile(
          "global_load_async_to_lds_b128 %0, %1, off\n\t"
          "global_load_async_to_lds_b128 %0, %1, off offset:16\n\t"
          "global_load_async_to_lds_b128 %0, %1, off offset:32\n\t"
          "global_load_async_to_lds_b128 %0, %1, off offset:48"
          :: "v"(ldst), "v"(gsrc) : "memory");
      __builtin_prefetch(wrow + kbn, 0, 1);   // global_prefetch_b8: weights one chunk ahead
    }

    const bf16_t* sbuf = (ch & 1) ? &sA[1][0] : &sA[0][0];
    const int kbase = ch * KCHUNK;

    for (int kc = 0; kc < KCHUNK; kc += 32) {
      v16bf a0, a1, bfrag;
#pragma unroll
      for (int v = 0; v < 8; ++v) {
        // A 16x32 bf16 layout: VGPR v holds K = base+2v(,+1); lanes16-31 shifted by 8
        const int ka = ((v < 4) ? (2 * v) : (16 + 2 * (v - 4))) + half * 8;
        a0[2 * v]     = sbuf[col * KCHUNK + kc + ka];
        a0[2 * v + 1] = sbuf[col * KCHUNK + kc + ka + 1];
        a1[2 * v]     = sbuf[(16 + col) * KCHUNK + kc + ka];
        a1[2 * v + 1] = sbuf[(16 + col) * KCHUNK + kc + ka + 1];
        // B 32x16 bf16 layout: same K order as A (VGPR v K = base+2v(,+1); lanes16-31 shifted by 8)
        const int kb = ka;
        bfrag[2 * v]     = wrow[kbase + kc + kb];
        bfrag[2 * v + 1] = wrow[kbase + kc + kb + 1];
      }
      c0 = __builtin_amdgcn_wmma_f32_16x16x32_bf16(false, a0, false, bfrag, (short)0, c0, false, false);
      c1 = __builtin_amdgcn_wmma_f32_16x16x32_bf16(false, a1, false, bfrag, (short)0, c1, false, false);
    }
    // all waves done reading this buffer before it can be re-staged
    __syncthreads();
  }

  // C layout: VGPR i -> lanes0-15 M=i, lanes16-31 M=8+i
#pragma unroll
  for (int i = 0; i < 8; ++i) {
    const int m = i + half * 8;
    out[(size_t)m * NG + n0 + col]        = c0[i];
    out[(size_t)(16 + m) * NG + n0 + col] = c1[i];
  }
}

// LSTM pointwise: c' = sig(f)*c + sig(i)*tanh(g); h' = sig(o)*tanh(c')
__global__ void lstm_pw_kernel(const float* __restrict__ gates,
                               float* __restrict__ c, float* __restrict__ h,
                               bf16_t* __restrict__ slot0, int ld0,
                               bf16_t* __restrict__ slot1, int ld1) {
  const int idx = blockIdx.x * blockDim.x + threadIdx.x;   // 0..BB*RNN-1
  const int b = idx >> 10, n = idx & 1023;
  const float* g = gates + (size_t)b * NG;
  const float gi = sigmoidf_(g[n]);
  const float gf = sigmoidf_(g[RNN + n]);
  const float gg = tanhf(g[2 * RNN + n]);
  const float go = sigmoidf_(g[3 * RNN + n]);
  const float c2 = gf * c[idx] + gi * gg;
  const float h2 = go * tanhf(c2);
  c[idx] = c2;
  h[idx] = h2;
  const bf16_t hb = (bf16_t)h2;
  slot0[(size_t)b * ld0 + n] = hb;
  if (slot1) slot1[(size_t)b * ld1 + n] = hb;
}

// location-sensitive attention step
__global__ void attention_step_kernel(
    const float* __restrict__ ah, const float* __restrict__ memory,
    const float* __restrict__ keys, const int* __restrict__ mem_len,
    const float* __restrict__ wq, const float* __restrict__ conv_w,
    const float* __restrict__ conv_b, const float* __restrict__ wloc,
    const float* __restrict__ wv,
    float* __restrict__ aw, float* __restrict__ awc, float* __restrict__ ctx,
    bf16_t* __restrict__ ctx_slot_att, bf16_t* __restrict__ ctx_slot_dec,
    float* __restrict__ align_out, int tstep) {
  const int b = blockIdx.x;
  const int t = threadIdx.x;                 // 0..T_ENC-1
  __shared__ float s_prev[T_ENC], s_cum[T_ENC], s_q[ATT_DIM], s_red[T_ENC], s_w[T_ENC];

  s_prev[t] = aw[b * T_ENC + t];
  s_cum[t]  = awc[b * T_ENC + t];
  __syncthreads();

  if (t < ATT_DIM) {           // q = ah @ wq^T
    const float* hv = ah + (size_t)b * RNN;
    const float* wr = wq + (size_t)t * RNN;
    float acc = 0.0f;
    for (int k = 0; k < RNN; ++k) acc += hv[k] * wr[k];
    s_q[t] = acc;
  }

  // 31-tap conv over [prev, cumulative] attention weights, 32 filters
  float locf[N_FILT];
#pragma unroll
  for (int f = 0; f < N_FILT; ++f) locf[f] = conv_b[f];
  for (int k = 0; k < KSIZE; ++k) {
    const int tp = t + k - (KSIZE - 1) / 2;
    if (tp >= 0 && tp < T_ENC) {
      const float a0 = s_prev[tp], a1 = s_cum[tp];
#pragma unroll
      for (int f = 0; f < N_FILT; ++f)
        locf[f] += conv_w[(f * 2 + 0) * KSIZE + k] * a0 +
                   conv_w[(f * 2 + 1) * KSIZE + k] * a1;
    }
  }
  __syncthreads();

  // energy = wv . tanh(q + key + wloc@locf)
  const float* krow = keys + ((size_t)b * T_ENC + t) * ATT_DIM;
  float e = 0.0f;
  for (int d = 0; d < ATT_DIM; ++d) {
    const float* wl = wloc + d * N_FILT;
    float lp = 0.0f;
#pragma unroll
    for (int f = 0; f < N_FILT; ++f) lp += wl[f] * locf[f];
    e += wv[d] * tanhf(s_q[d] + krow[d] + lp);
  }
  if (t >= mem_len[b]) e = -1.0e9f;

  // softmax over T_ENC
  s_red[t] = e; __syncthreads();
  for (int s = T_ENC / 2; s > 0; s >>= 1) {
    if (t < s) s_red[t] = fmaxf(s_red[t], s_red[t + s]);
    __syncthreads();
  }
  const float mx = s_red[0];
  __syncthreads();
  const float ex = expf(e - mx);
  s_red[t] = ex; __syncthreads();
  for (int s = T_ENC / 2; s > 0; s >>= 1) {
    if (t < s) s_red[t] += s_red[t + s];
    __syncthreads();
  }
  const float w = ex / s_red[0];
  s_w[t] = w;
  aw[b * T_ENC + t]  = w;
  awc[b * T_ENC + t] = s_cum[t] + w;
  align_out[((size_t)b * T_OUT + tstep) * T_ENC + t] = w;
  __syncthreads();

  // ctx = w @ memory  (coalesced over embedding dim)
  for (int e0 = t; e0 < EMB; e0 += T_ENC) {
    const float* mb = memory + ((size_t)b * T_ENC) * EMB + e0;
    float acc = 0.0f;
    for (int tt = 0; tt < T_ENC; ++tt) acc += s_w[tt] * mb[(size_t)tt * EMB];
    ctx[b * EMB + e0] = acc;
    const bf16_t cb = (bf16_t)acc;
    ctx_slot_att[(size_t)b * KATT + e0] = cb;
    ctx_slot_dec[(size_t)b * KDEC + e0] = cb;
  }
}

// mel/gate projection for this step
__global__ void output_step_kernel(const float* __restrict__ dh, const float* __restrict__ ctx,
                                   const float* __restrict__ wproj, const float* __restrict__ bproj,
                                   const float* __restrict__ wgate, const float* __restrict__ bgate,
                                   float* __restrict__ mel_out, float* __restrict__ gate_out,
                                   int tstep) {
  const int b = blockIdx.x;
  const int j = threadIdx.x;    // 0..127 (80 mel + 1 gate used)
  const float* h = dh + (size_t)b * RNN;
  const float* c = ctx + (size_t)b * EMB;
  if (j < N_MEL) {
    const float* wr = wproj + (size_t)j * (RNN + EMB);
    float acc = bproj[j];
    for (int k = 0; k < RNN; ++k) acc += h[k] * wr[k];
    for (int k = 0; k < EMB; ++k) acc += c[k] * wr[RNN + k];
    mel_out[((size_t)b * N_MEL + j) * T_OUT + tstep] = acc;
  } else if (j == N_MEL) {
    float acc = bgate[0];
    for (int k = 0; k < RNN; ++k) acc += h[k] * wgate[k];
    for (int k = 0; k < EMB; ++k) acc += c[k] * wgate[RNN + k];
    gate_out[(size_t)b * T_OUT + tstep] = acc;
  }
}

// ============================================================
// host launcher
// ============================================================
extern "C" void kernel_launch(void* const* d_in, const int* in_sizes, int n_in,
                              void* d_out, int out_size, void* d_ws, size_t ws_size,
                              hipStream_t stream) {
  (void)in_sizes; (void)n_in; (void)out_size; (void)ws_size;
  const float* memory     = (const float*)d_in[0];
  const float* dec_inputs = (const float*)d_in[1];
  const int*   mem_len    = (const int*)  d_in[2];
  const float* prenet_w1  = (const float*)d_in[3];
  const float* prenet_w2  = (const float*)d_in[4];
  const float* att_wih    = (const float*)d_in[5];
  const float* att_whh    = (const float*)d_in[6];
  const float* att_bih    = (const float*)d_in[7];
  const float* att_bhh    = (const float*)d_in[8];
  const float* wq         = (const float*)d_in[9];
  const float* wk         = (const float*)d_in[10];
  const float* conv_w     = (const float*)d_in[11];
  const float* conv_b     = (const float*)d_in[12];
  const float* wloc       = (const float*)d_in[13];
  const float* wv         = (const float*)d_in[14];
  const float* dec_wih    = (const float*)d_in[15];
  const float* dec_whh    = (const float*)d_in[16];
  const float* dec_bih    = (const float*)d_in[17];
  const float* dec_bhh    = (const float*)d_in[18];
  const float* wproj      = (const float*)d_in[19];
  const float* bproj      = (const float*)d_in[20];
  const float* wgate      = (const float*)d_in[21];
  const float* bgate      = (const float*)d_in[22];

  // output sections (reference return order: mel, gate, alignments)
  float* out       = (float*)d_out;
  float* mel_out   = out;
  float* gate_out  = out + (size_t)BB * N_MEL * T_OUT;
  float* align_out = gate_out + (size_t)BB * T_OUT;

  // ---- workspace layout ----
  char* ws = (char*)d_ws;
  size_t off = 0;
  auto alloc = [&](size_t bytes) -> char* {
    char* p = ws + off;
    off += (bytes + 255) & ~(size_t)255;
    return p;
  };
  bf16_t* W_att  = (bf16_t*)alloc((size_t)NG * KATT * sizeof(bf16_t));
  bf16_t* W_dec  = (bf16_t*)alloc((size_t)NG * KDEC * sizeof(bf16_t));
  float*  b_att  = (float*) alloc(NG * sizeof(float));
  float*  b_dec  = (float*) alloc(NG * sizeof(float));
  bf16_t* xs_bf  = (bf16_t*)alloc((size_t)T_OUT * BB * PRENETD * sizeof(bf16_t));
  float*  keys   = (float*) alloc((size_t)BB * T_ENC * ATT_DIM * sizeof(float));
  float*  gates  = (float*) alloc((size_t)BB * NG * sizeof(float));
  // mutable per-step state (zeroed every launch for deterministic replay)
  size_t state_begin = off;
  float*  ah    = (float*) alloc((size_t)BB * RNN * sizeof(float));
  float*  ac    = (float*) alloc((size_t)BB * RNN * sizeof(float));
  float*  dh    = (float*) alloc((size_t)BB * RNN * sizeof(float));
  float*  dc    = (float*) alloc((size_t)BB * RNN * sizeof(float));
  float*  ctx   = (float*) alloc((size_t)BB * EMB * sizeof(float));
  float*  aw    = (float*) alloc((size_t)BB * T_ENC * sizeof(float));
  float*  awc   = (float*) alloc((size_t)BB * T_ENC * sizeof(float));
  bf16_t* A_att = (bf16_t*)alloc((size_t)BB * KATT * sizeof(bf16_t));
  bf16_t* A_dec = (bf16_t*)alloc((size_t)BB * KDEC * sizeof(bf16_t));
  size_t state_words = (off - state_begin) / 4;

  // ---- phase A: one-shot parallel prep ----
  zero_words_kernel<<<(unsigned)((state_words + 255) / 256), 256, 0, stream>>>(
      (unsigned int*)(ws + state_begin), state_words);
  prep_w_kernel<<<(unsigned)(((size_t)NG * KATT + 255) / 256), 256, 0, stream>>>(
      att_wih, att_whh, att_bih, att_bhh, W_att, b_att, PRENETD + EMB, KATT);
  prep_w_kernel<<<(unsigned)(((size_t)NG * KDEC + 255) / 256), 256, 0, stream>>>(
      dec_wih, dec_whh, dec_bih, dec_bhh, W_dec, b_dec, RNN + EMB, KDEC);
  prenet_kernel<<<T_OUT * BB, PRENETD, 0, stream>>>(dec_inputs, prenet_w1, prenet_w2, xs_bf);
  keys_kernel<<<BB * T_ENC, ATT_DIM, 0, stream>>>(memory, wk, keys);

  // activation-panel slot pointers
  bf16_t* ctx_slot_att = A_att + PRENETD;        // A_att[:, 256:768]
  bf16_t* ah_slot_att  = A_att + PRENETD + EMB;  // A_att[:, 768:1792]
  bf16_t* ah_slot_dec  = A_dec;                  // A_dec[:, 0:1024]
  bf16_t* ctx_slot_dec = A_dec + RNN;            // A_dec[:, 1024:1536]
  bf16_t* dh_slot_dec  = A_dec + RNN + EMB;      // A_dec[:, 1536:2560]

  // ---- phase B: 400 sequential decoder steps ----
  for (int t = 0; t < T_OUT; ++t) {
    copy_x_kernel<<<(BB * PRENETD) / 256, 256, 0, stream>>>(xs_bf, A_att, t);
    gemm_gates_kernel<<<NG / 128, 256, 0, stream>>>(A_att, W_att, b_att, gates, KATT);
    lstm_pw_kernel<<<(BB * RNN) / 256, 256, 0, stream>>>(
        gates, ac, ah, ah_slot_att, KATT, ah_slot_dec, KDEC);
    attention_step_kernel<<<BB, T_ENC, 0, stream>>>(
        ah, memory, keys, mem_len, wq, conv_w, conv_b, wloc, wv,
        aw, awc, ctx, ctx_slot_att, ctx_slot_dec, align_out, t);
    gemm_gates_kernel<<<NG / 128, 256, 0, stream>>>(A_dec, W_dec, b_dec, gates, KDEC);
    lstm_pw_kernel<<<(BB * RNN) / 256, 256, 0, stream>>>(
        gates, dc, dh, dh_slot_dec, KDEC, (bf16_t*)nullptr, 0);
    output_step_kernel<<<BB, 128, 0, stream>>>(
        dh, ctx, wproj, bproj, wgate, bgate, mel_out, gate_out, t);
  }
}
